// VisionMamba_slowfast_54966991454782
// MI455X (gfx1250) — matrix operand order repaired and checked by reading, not compiled, not run on hardware
//
#include <hip/hip_runtime.h>
#include <hip/hip_bf16.h>

// ---------------------------------------------------------------------------
// VisionMamba slow/fast forward for gfx1250 (MI455X).
// Dense projections: v_wmma_f32_16x16x32_f16, wave32, 32x32 macro-tile per
// wave (2x2 WMMA register tiling -> 4 WMMAs per 16 b128 loads).  Edge tiles
// use a branchless clamped-address path; full tiles use unguarded stores.
// ---------------------------------------------------------------------------

typedef __attribute__((ext_vector_type(16))) _Float16 v16h;
typedef __attribute__((ext_vector_type(8)))  float    v8f;
typedef __attribute__((ext_vector_type(4)))  float    f4v;

#define EPS 1e-5f

static inline int gsz(long long n, int b) { return (int)((n + b - 1) / b); }

// A fragment (16-bit A 16x32 layout): p -> floats p[0..7], p[16..23]
__device__ __forceinline__ v16h ld_afrag(const float* p)
{
  f4v a0 = *(const f4v*)(p);
  f4v a1 = *(const f4v*)(p + 4);
  f4v a2 = *(const f4v*)(p + 16);
  f4v a3 = *(const f4v*)(p + 20);
  v16h r;
  r[0]=(_Float16)a0.x;  r[1]=(_Float16)a0.y;  r[2]=(_Float16)a0.z;  r[3]=(_Float16)a0.w;
  r[4]=(_Float16)a1.x;  r[5]=(_Float16)a1.y;  r[6]=(_Float16)a1.z;  r[7]=(_Float16)a1.w;
  r[8]=(_Float16)a2.x;  r[9]=(_Float16)a2.y;  r[10]=(_Float16)a2.z; r[11]=(_Float16)a2.w;
  r[12]=(_Float16)a3.x; r[13]=(_Float16)a3.y; r[14]=(_Float16)a3.z; r[15]=(_Float16)a3.w;
  return r;
}

// B fragment (32x16 layout): p -> floats p[0..7], p[16..23]
__device__ __forceinline__ v16h ld_bfrag(const float* p)
{
  f4v b0 = *(const f4v*)(p);
  f4v b1 = *(const f4v*)(p + 4);
  f4v b2 = *(const f4v*)(p + 16);
  f4v b3 = *(const f4v*)(p + 20);
  v16h r;
  r[0]=(_Float16)b0.x;  r[1]=(_Float16)b0.y;  r[2]=(_Float16)b0.z;  r[3]=(_Float16)b0.w;
  r[4]=(_Float16)b1.x;  r[5]=(_Float16)b1.y;  r[6]=(_Float16)b1.z;  r[7]=(_Float16)b1.w;
  r[8]=(_Float16)b2.x;  r[9]=(_Float16)b2.y;  r[10]=(_Float16)b2.z; r[11]=(_Float16)b2.w;
  r[12]=(_Float16)b3.x; r[13]=(_Float16)b3.y; r[14]=(_Float16)b3.z; r[15]=(_Float16)b3.w;
  return r;
}

// guarded store of one 16x16 tile (C/D layout: vgpr v -> M = v + 8*hi)
__device__ __forceinline__ void st_tile(float* __restrict__ C, int ldc,
                                        const float* __restrict__ bias,
                                        const v8f& acc, int sm, int sn,
                                        int row, int hi, int M, int N)
{
  const int col = sn + row;
  if (col < N) {
    const float bv = bias ? bias[col] : 0.0f;
#pragma unroll
    for (int v = 0; v < 8; ++v) {
      int r = sm + hi * 8 + v;
      if (r < M) C[(size_t)r * ldc + col] = acc[v] + bv;
    }
  }
}

// unguarded store for full in-bounds tiles (straight-line b32 stores)
__device__ __forceinline__ void st_tile_fast(float* __restrict__ C, int ldc,
                                             const float* __restrict__ bias,
                                             const v8f& acc, int sm, int sn,
                                             int row, int hi)
{
  const int col = sn + row;
  const float bv = bias ? bias[col] : 0.0f;
  float* cp = C + (size_t)(sm + hi * 8) * ldc + col;
#pragma unroll
  for (int v = 0; v < 8; ++v) cp[(size_t)v * ldc] = acc[v] + bv;
}

// branchless 16x16 edge tile (clamped addresses, value-or-zero selects)
__device__ __forceinline__ void gemm_tile_edge(const float* __restrict__ A, int lda,
                                               const float* __restrict__ W, int ldw,
                                               const float* __restrict__ bias,
                                               float* __restrict__ C, int ldc,
                                               int M, int N, int K,
                                               int sm, int sn, int row, int hi)
{
  const int  ar   = (sm + row < M) ? (sm + row) : (M - 1);
  const int  br   = (sn + row < N) ? (sn + row) : (N - 1);
  const bool aval = (sm + row) < M;
  const bool bval = (sn + row) < N;
  const float* Arow = A + (size_t)ar * lda;
  const float* Wrow = W + (size_t)br * ldw;
  v8f acc = {};
  for (int k0 = 0; k0 < K; k0 += 32) {
    v16h af, bf;
    const int ka = k0 + hi * 8;
#pragma unroll
    for (int j = 0; j < 8; ++j) {
      int k1 = ka + j, k2 = ka + 16 + j;
      float v1 = Arow[(k1 < K) ? k1 : (K - 1)];
      float v2 = Arow[(k2 < K) ? k2 : (K - 1)];
      af[j]     = (aval && k1 < K) ? (_Float16)v1 : (_Float16)0.0f;
      af[j + 8] = (aval && k2 < K) ? (_Float16)v2 : (_Float16)0.0f;
    }
    const int kb = k0 + hi * 8;
#pragma unroll
    for (int j = 0; j < 16; ++j) {
      int k1 = kb + (j < 8 ? j : j + 8);
      float v1 = Wrow[(k1 < K) ? k1 : (K - 1)];
      bf[j] = (bval && k1 < K) ? (_Float16)v1 : (_Float16)0.0f;
    }
    acc = __builtin_amdgcn_wmma_f32_16x16x32_f16(false, af, false, bf,
                                                 (short)0, acc, false, false);
  }
  st_tile(C, ldc, bias, acc, sm, sn, row, hi, M, N);
}

// ------------------------- WMMA GEMM: C = A @ W^T (+bias) ------------------
// A: [M,K] row stride lda (fp32).  W: [N,K] row stride ldw (fp32, "B^T").
// One wave computes a 32x32 macro-tile (2x2 WMMAs); grid over macro-tiles.
__global__ __launch_bounds__(32)
void k_gemm_wmma(const float* __restrict__ A, int lda,
                 const float* __restrict__ W, int ldw,
                 const float* __restrict__ bias,
                 float* __restrict__ C, int ldc,
                 int M, int N, int K)
{
  const int tilesN = (N + 31) >> 5;
  const int tile = blockIdx.x;
  const int tm = (tile / tilesN) << 5;
  const int tn = (tile % tilesN) << 5;
  const int lane = threadIdx.x & 31;
  const int row = lane & 15;
  const int hi  = lane >> 4;

  const bool fast = (tm + 32 <= M) && (tn + 32 <= N) && ((K & 31) == 0) &&
                    ((lda & 3) == 0) && ((ldw & 3) == 0);   // wave-uniform

  if (fast) {
    v8f acc00 = {}, acc01 = {}, acc10 = {}, acc11 = {};
    const float* Ar0 = A + (size_t)(tm + row) * lda;
    const float* Ar1 = Ar0 + (size_t)16 * lda;
    const float* Wr0 = W + (size_t)(tn + row) * ldw;
    const float* Wr1 = Wr0 + (size_t)16 * ldw;
    for (int k0 = 0; k0 < K; k0 += 32) {
      const int oa = k0 + hi * 8;
      const int ob = k0 + hi * 8;
      v16h a0 = ld_afrag(Ar0 + oa);
      v16h a1 = ld_afrag(Ar1 + oa);
      v16h b0 = ld_bfrag(Wr0 + ob);
      v16h b1 = ld_bfrag(Wr1 + ob);
      if (k0 + 32 < K) {               // prefetch next K tile (global_prefetch_b8)
        __builtin_prefetch(Ar0 + oa + 32, 0, 0);
        __builtin_prefetch(Ar1 + oa + 32, 0, 0);
        __builtin_prefetch(Wr0 + ob + 32, 0, 0);
        __builtin_prefetch(Wr1 + ob + 32, 0, 0);
      }
      acc00 = __builtin_amdgcn_wmma_f32_16x16x32_f16(false, a0, false, b0, (short)0, acc00, false, false);
      acc01 = __builtin_amdgcn_wmma_f32_16x16x32_f16(false, a0, false, b1, (short)0, acc01, false, false);
      acc10 = __builtin_amdgcn_wmma_f32_16x16x32_f16(false, a1, false, b0, (short)0, acc10, false, false);
      acc11 = __builtin_amdgcn_wmma_f32_16x16x32_f16(false, a1, false, b1, (short)0, acc11, false, false);
    }
    st_tile_fast(C, ldc, bias, acc00, tm,      tn,      row, hi);
    st_tile_fast(C, ldc, bias, acc01, tm,      tn + 16, row, hi);
    st_tile_fast(C, ldc, bias, acc10, tm + 16, tn,      row, hi);
    st_tile_fast(C, ldc, bias, acc11, tm + 16, tn + 16, row, hi);
  } else {
    // up to four 16x16 sub-tiles, each via the branchless edge path
#pragma unroll
    for (int st = 0; st < 4; ++st) {
      int sm = tm + ((st >> 1) << 4);
      int sn = tn + ((st & 1) << 4);
      if (sm >= M || sn >= N) continue;
      gemm_tile_edge(A, lda, W, ldw, bias, C, ldc, M, N, K, sm, sn, row, hi);
    }
  }
}

// ------------------------- elementwise helpers -----------------------------
__global__ void k_add(const float* __restrict__ a, const float* __restrict__ b,
                      float* __restrict__ o, long long n)
{
  long long i = (long long)blockIdx.x * blockDim.x + threadIdx.x;
  if (i < n) o[i] = a[i] + b[i];
}

// rmsnorm over last dim; optional second input added first (for h + residual)
__global__ __launch_bounds__(32)
void k_rmsnorm(const float* __restrict__ X, const float* __restrict__ X2,
               const float* __restrict__ w, float* __restrict__ out,
               int ntok, int d)
{
  const int tok = blockIdx.x;
  const int lane = threadIdx.x;
  const float* xp  = X + (size_t)tok * d;
  const float* xp2 = X2 ? X2 + (size_t)tok * d : nullptr;
  float ss = 0.0f;
  for (int c = lane; c < d; c += 32) {
    float v = xp[c] + (xp2 ? xp2[c] : 0.0f);
    ss += v * v;
  }
  for (int off = 16; off > 0; off >>= 1) ss += __shfl_xor(ss, off, 32);
  const float inv = rsqrtf(ss / (float)d + EPS);
  float* op = out + (size_t)tok * d;
  for (int c = lane; c < d; c += 32) {
    float v = xp[c] + (xp2 ? xp2[c] : 0.0f);
    op[c] = v * inv * w[c];
  }
}

// causal depthwise conv (K=4) + SiLU; optional sequence reversal (rev)
__global__ void k_dwconv_silu(const float* __restrict__ X, int ldx,
                              const float* __restrict__ w,     // [di,4]
                              const float* __restrict__ bc,    // [di]
                              float* __restrict__ out,         // [B,L,di]
                              int Bn, int L, int di, int rev)
{
  long long idx = (long long)blockIdx.x * blockDim.x + threadIdx.x;
  long long total = (long long)Bn * L * di;
  if (idx >= total) return;
  int c = (int)(idx % di);
  int l = (int)((idx / di) % L);
  int b = (int)(idx / ((long long)di * L));
  float acc = bc[c];
#pragma unroll
  for (int k = 0; k < 4; ++k) {
    int ls = l - 3 + k;
    if (ls < 0) continue;
    int lx = rev ? (L - 1 - ls) : ls;
    acc += X[((size_t)b * L + lx) * ldx + c] * w[c * 4 + k];
  }
  out[idx] = acc / (1.0f + expf(-acc));       // silu
}

__global__ void k_softplus(float* __restrict__ v, const float* __restrict__ bias,
                           long long n, int di)
{
  long long i = (long long)blockIdx.x * blockDim.x + threadIdx.x;
  if (i >= n) return;
  float t = v[i] + bias[i % di];
  v[i] = (t > 20.0f) ? t : log1pf(expf(t));
}

// selective scan: one thread per (batch, channel), 16 states in registers
__global__ void k_scan(const float* __restrict__ u,      // [B,L,di]
                       const float* __restrict__ delta,  // [B,L,di]
                       const float* __restrict__ Alog,   // [di,16]
                       const float* __restrict__ dbl,    // [B,L,lddbl]; Bm at +offB, Cm at +offB+16
                       int lddbl, int offB,
                       const float* __restrict__ Dp,     // [di]
                       float* __restrict__ ys,           // [B,L,di]
                       int Bn, int L, int di)
{
  int idx = blockIdx.x * blockDim.x + threadIdx.x;
  if (idx >= Bn * di) return;
  int d = idx % di;
  int b = idx / di;
  float Ar[16], h[16];
#pragma unroll
  for (int n = 0; n < 16; ++n) { Ar[n] = -expf(Alog[d * 16 + n]); h[n] = 0.0f; }
  const float Dd = Dp[d];
  for (int l = 0; l < L; ++l) {
    size_t tix = (size_t)b * L + l;
    float dt = delta[tix * di + d];
    float ut = u[tix * di + d];
    const float* bcm = dbl + tix * lddbl + offB;
    float y = 0.0f;
#pragma unroll
    for (int n = 0; n < 16; ++n) {
      float dA = expf(dt * Ar[n]);
      h[n] = dA * h[n] + dt * bcm[n] * ut;
      y += h[n] * bcm[16 + n];
    }
    ys[tix * di + d] = y + ut * Dd;
  }
}

// y = (ys_fwd[l] + ys_bwd[L-1-l]) * silu(z),  z lives at xz[..., di:2di]
__global__ void k_combine(const float* __restrict__ ysf, const float* __restrict__ ysb,
                          const float* __restrict__ xz, int ldxz, int zoff,
                          float* __restrict__ y, int Bn, int L, int di)
{
  long long idx = (long long)blockIdx.x * blockDim.x + threadIdx.x;
  long long total = (long long)Bn * L * di;
  if (idx >= total) return;
  int c = (int)(idx % di);
  int l = (int)((idx / di) % L);
  int b = (int)(idx / ((long long)di * L));
  float zf = xz[((size_t)b * L + l) * ldxz + zoff + c];
  float s  = zf / (1.0f + expf(-zf));
  float v  = ysf[idx] + ysb[((size_t)b * L + (L - 1 - l)) * di + c];
  y[idx] = v * s;
}

// fused patch-embed conv3d (VALID, stride (stT,16,16)) + tokenize + pos/tpos
__global__ void k_patch_embed(const float* __restrict__ X,   // [B,3,T,224,224]
                              const float* __restrict__ W,   // [Co,3,Kt,16,16]
                              const float* __restrict__ bias,// [Co]
                              const float* __restrict__ pos, // [196,Co]
                              const float* __restrict__ tpos,// [Tt,Co]
                              float* __restrict__ tok,       // [B,Tt*196,Co]
                              int Bn, int T, int Tt, int Kt, int stT, int Co)
{
  long long idx = (long long)blockIdx.x * blockDim.x + threadIdx.x;
  long long total = (long long)Bn * Tt * 196 * Co;
  if (idx >= total) return;
  int co = (int)(idx % Co);
  int hw = (int)((idx / Co) % 196);
  int to = (int)((idx / ((long long)Co * 196)) % Tt);
  int b  = (int)(idx / ((long long)Co * 196 * Tt));
  int h = hw / 14, w = hw % 14;
  float acc = bias[co];
  for (int ci = 0; ci < 3; ++ci)
    for (int kt = 0; kt < Kt; ++kt) {
      int tin = to * stT + kt;
      const float* xp = X + (((size_t)(b * 3 + ci) * T + tin) * 224 + h * 16) * 224 + w * 16;
      const float* wp = W + ((size_t)(co * 3 + ci) * Kt + kt) * 256;
      for (int ph = 0; ph < 16; ++ph)
#pragma unroll
        for (int pw = 0; pw < 16; ++pw)
          acc += xp[ph * 224 + pw] * wp[ph * 16 + pw];
    }
  tok[idx] = acc + pos[hw * Co + co] + tpos[to * Co + co];
}

// lateral: conv3d(fast tokens, (192,96,3,1,1), stride (2,1,1), tpad 1)
//          + BN + ReLU + slow tokens; output in token layout [B,784,192]
__global__ void k_lateral(const float* __restrict__ hs,   // [B,784,192]
                          const float* __restrict__ ff,   // [B,1568,96]
                          const float* __restrict__ W,    // [192,96,3]
                          const float* __restrict__ bias,
                          const float* __restrict__ bn_g, const float* __restrict__ bn_b,
                          const float* __restrict__ bn_m, const float* __restrict__ bn_v,
                          float* __restrict__ out, int Bn)
{
  long long idx = (long long)blockIdx.x * blockDim.x + threadIdx.x;
  long long total = (long long)Bn * 784 * 192;
  if (idx >= total) return;
  int co = (int)(idx % 192);
  int pos = (int)((idx / 192) % 784);
  int b = (int)(idx / (192LL * 784));
  int to = pos / 196, hw = pos % 196;
  float acc = bias[co];
  for (int kt = 0; kt < 3; ++kt) {
    int tin = to * 2 - 1 + kt;
    if (tin < 0 || tin >= 8) continue;
    const float* fp = ff + ((size_t)b * 1568 + tin * 196 + hw) * 96;
    const float* wp = W + (size_t)co * 96 * 3 + kt;
    for (int ci = 0; ci < 96; ++ci) acc += fp[ci] * wp[ci * 3];
  }
  float scale = bn_g[co] * rsqrtf(bn_v[co] + EPS);
  float y = (acc - bn_m[co]) * scale + bn_b[co];
  y = fmaxf(y, 0.0f);
  out[idx] = y + hs[idx];
}

// upsample: concat([fast(96ch), repeat(slow,2,t)(192ch)]) -> conv (192,288,3,1,1)
// output token layout [B,1568,192]
__global__ void k_upconv(const float* __restrict__ ff,  // [B,1568,96]
                         const float* __restrict__ sf,  // [B,784,192]
                         const float* __restrict__ W,   // [192,288,3]
                         const float* __restrict__ bias,
                         float* __restrict__ out, int Bn)
{
  long long idx = (long long)blockIdx.x * blockDim.x + threadIdx.x;
  long long total = (long long)Bn * 1568 * 192;
  if (idx >= total) return;
  int co = (int)(idx % 192);
  int pos = (int)((idx / 192) % 1568);
  int b = (int)(idx / (192LL * 1568));
  int t = pos / 196, hw = pos % 196;
  float acc = bias[co];
  for (int kt = 0; kt < 3; ++kt) {
    int tin = t - 1 + kt;
    if (tin < 0 || tin >= 8) continue;
    const float* fp = ff + ((size_t)b * 1568 + tin * 196 + hw) * 96;
    const float* sp = sf + ((size_t)b * 784 + (tin >> 1) * 196 + hw) * 192;
    const float* wp = W + (size_t)co * 288 * 3 + kt;
    for (int ci = 0; ci < 96;  ++ci) acc += fp[ci] * wp[ci * 3];
    for (int ci = 0; ci < 192; ++ci) acc += sp[ci] * wp[(96 + ci) * 3];
  }
  out[idx] = acc;
}

__global__ void k_meanpool(const float* __restrict__ tok, float* __restrict__ out,
                           int Bn, int Lp, int Dc)
{
  int idx = blockIdx.x * blockDim.x + threadIdx.x;
  if (idx >= Bn * Dc) return;
  int c = idx % Dc, b = idx / Dc;
  float s = 0.0f;
  for (int l = 0; l < Lp; ++l) s += tok[((size_t)b * Lp + l) * Dc + c];
  out[idx] = s / (float)Lp;
}

// ------------------------- host orchestration ------------------------------

struct StackW {
  const float *norm_w, *in_proj, *out_proj;
  const float *conv_w[2], *conv_b[2], *x_proj[2], *dt_w[2], *dt_b[2], *A_log[2], *Dp[2];
  int d, di, dtr, dblN, L;
};
struct Bufs { float *norm, *xz, *xc[2], *dbl[2], *delta[2], *ys[2], *y; };

static inline int gtiles(int M, int N) { return gsz(M, 32) * gsz(N, 32); }

static void run_stack(const StackW& S, int l0, int l1, int Bn,
                      float* hidden, float* resid, const Bufs& B, hipStream_t stream)
{
  const int L = S.L, d = S.d, di = S.di, dtr = S.dtr, dblN = S.dblN;
  const int M = Bn * L;
  const int EB = 256;
  for (int i = l0; i < l1; ++i) {
    // residual += hidden
    k_add<<<gsz((long long)M * d, EB), EB, 0, stream>>>(resid, hidden, resid, (long long)M * d);
    // normed = rmsnorm(residual)
    k_rmsnorm<<<M, 32, 0, stream>>>(resid, nullptr, S.norm_w + (size_t)i * d, B.norm, M, d);
    // xz = normed @ in_proj^T                 [M, 2di]
    k_gemm_wmma<<<gtiles(M, 2 * di), 32, 0, stream>>>(
        B.norm, d, S.in_proj + (size_t)i * 2 * di * d, d, nullptr, B.xz, 2 * di, M, 2 * di, d);
    for (int r = 0; r < 2; ++r) {
      k_dwconv_silu<<<gsz((long long)M * di, EB), EB, 0, stream>>>(
          B.xz, 2 * di, S.conv_w[r] + (size_t)i * di * 4, S.conv_b[r] + (size_t)i * di,
          B.xc[r], Bn, L, di, r);
      // dbl = xc @ x_proj^T                   [M, dblN]
      k_gemm_wmma<<<gtiles(M, dblN), 32, 0, stream>>>(
          B.xc[r], di, S.x_proj[r] + (size_t)i * dblN * di, di, nullptr, B.dbl[r], dblN, M, dblN, di);
      // delta = dt @ dt_w^T                   [M, di]  (dt = dbl[:, :dtr])
      k_gemm_wmma<<<gtiles(M, di), 32, 0, stream>>>(
          B.dbl[r], dblN, S.dt_w[r] + (size_t)i * di * dtr, dtr, nullptr, B.delta[r], di, M, di, dtr);
      k_softplus<<<gsz((long long)M * di, EB), EB, 0, stream>>>(
          B.delta[r], S.dt_b[r] + (size_t)i * di, (long long)M * di, di);
      k_scan<<<gsz(Bn * di, 64), 64, 0, stream>>>(
          B.xc[r], B.delta[r], S.A_log[r] + (size_t)i * di * 16,
          B.dbl[r], dblN, dtr, S.Dp[r] + (size_t)i * di, B.ys[r], Bn, L, di);
    }
    k_combine<<<gsz((long long)M * di, EB), EB, 0, stream>>>(
        B.ys[0], B.ys[1], B.xz, 2 * di, di, B.y, Bn, L, di);
    // hidden = y @ out_proj^T                 [M, d]
    k_gemm_wmma<<<gtiles(M, d), 32, 0, stream>>>(
        B.y, di, S.out_proj + (size_t)i * d * di, di, nullptr, hidden, d, M, d, di);
  }
}

static float* carve(char*& p, size_t n) { float* r = (float*)p; p += n * sizeof(float); return r; }

extern "C" void kernel_launch(void* const* d_in, const int* in_sizes, int n_in,
                              void* d_out, int out_size, void* d_ws, size_t ws_size,
                              hipStream_t stream)
{
  (void)in_sizes; (void)n_in; (void)out_size; (void)ws_size;
  // --- inputs (dict insertion-order flatten) ---
  const float* x           = (const float*)d_in[0];
  const float* ps_w        = (const float*)d_in[1];   // proj_slow_w
  const float* ps_b        = (const float*)d_in[2];
  const float* pf_w        = (const float*)d_in[3];   // proj_fast_w
  const float* pf_b        = (const float*)d_in[4];
  const float* pos_slow    = (const float*)d_in[5];
  const float* tpos_slow   = (const float*)d_in[6];
  const float* pos_fast    = (const float*)d_in[7];
  const float* tpos_fast   = (const float*)d_in[8];
  const float* norm_f_slow = (const float*)d_in[9];
  const float* norm_f_fast = (const float*)d_in[10];
  const float* head_w      = (const float*)d_in[11];
  const float* head_b      = (const float*)d_in[12];

  StackW SS{}, SF{};
  {  // slow_params: d_in[13..29], order per make_stack insertion
    SS.d = 192; SS.di = 384; SS.dtr = 12; SS.dblN = 12 + 32; SS.L = 784;
    SS.norm_w   = (const float*)d_in[13];
    SS.in_proj  = (const float*)d_in[14];
    SS.out_proj = (const float*)d_in[15];
    for (int r = 0; r < 2; ++r) {
      int o = 16 + 7 * r;
      SS.conv_w[r] = (const float*)d_in[o + 0];
      SS.conv_b[r] = (const float*)d_in[o + 1];
      SS.x_proj[r] = (const float*)d_in[o + 2];
      SS.dt_w[r]   = (const float*)d_in[o + 3];
      SS.dt_b[r]   = (const float*)d_in[o + 4];
      SS.A_log[r]  = (const float*)d_in[o + 5];
      SS.Dp[r]     = (const float*)d_in[o + 6];
    }
  }
  {  // fast_params: d_in[30..46]
    SF.d = 96; SF.di = 192; SF.dtr = 6; SF.dblN = 6 + 32; SF.L = 1568;
    SF.norm_w   = (const float*)d_in[30];
    SF.in_proj  = (const float*)d_in[31];
    SF.out_proj = (const float*)d_in[32];
    for (int r = 0; r < 2; ++r) {
      int o = 33 + 7 * r;
      SF.conv_w[r] = (const float*)d_in[o + 0];
      SF.conv_b[r] = (const float*)d_in[o + 1];
      SF.x_proj[r] = (const float*)d_in[o + 2];
      SF.dt_w[r]   = (const float*)d_in[o + 3];
      SF.dt_b[r]   = (const float*)d_in[o + 4];
      SF.A_log[r]  = (const float*)d_in[o + 5];
      SF.Dp[r]     = (const float*)d_in[o + 6];
    }
  }
  const float* lat_w   = (const float*)d_in[47];
  const float* lat_b   = (const float*)d_in[48];
  const float* lat_bng = (const float*)d_in[49];
  const float* lat_bnb = (const float*)d_in[50];
  const float* lat_bnm = (const float*)d_in[51];
  const float* lat_bnv = (const float*)d_in[52];
  const float* ups_w   = (const float*)d_in[53];
  const float* ups_b   = (const float*)d_in[54];

  const int Bn = 4;
  const int EB = 256;

  // --- workspace carve (~67 MB fp32) ---
  char* p = (char*)d_ws;
  const size_t TOK = 602112;        // = 4*784*192 = 4*1568*96
  float* tok_s    = carve(p, TOK);
  float* tok_f    = carve(p, TOK);
  float* resid    = carve(p, TOK);
  float* fastfeat = carve(p, TOK);
  float* slowfeat = carve(p, TOK);
  float* lat_out  = carve(p, TOK);
  Bufs Bf{};
  Bf.norm     = carve(p, TOK);
  Bf.xz       = carve(p, 2408448);  // M * 2di (same for both stacks)
  Bf.xc[0]    = carve(p, 1204224);
  Bf.xc[1]    = carve(p, 1204224);
  Bf.dbl[0]   = carve(p, 238336);   // max(4*1568*38, 4*784*44)
  Bf.dbl[1]   = carve(p, 238336);
  Bf.delta[0] = carve(p, 1204224);
  Bf.delta[1] = carve(p, 1204224);
  Bf.ys[0]    = carve(p, 1204224);
  Bf.ys[1]    = carve(p, 1204224);
  Bf.y        = carve(p, 1204224);
  float* up_tok = carve(p, 1204224);
  float* pooled = carve(p, 768);

  // --- patch embed + tokenize + positional ---
  k_patch_embed<<<gsz((long long)TOK, EB), EB, 0, stream>>>(
      x, ps_w, ps_b, pos_slow, tpos_slow, tok_s, Bn, 8, 4, 2, 2, 192);
  k_patch_embed<<<gsz((long long)TOK, EB), EB, 0, stream>>>(
      x, pf_w, pf_b, pos_fast, tpos_fast, tok_f, Bn, 8, 8, 1, 1, 96);

  // --- fast stack (24 layers), hidden lives in tok_f ---
  hipMemsetAsync(resid, 0, TOK * sizeof(float), stream);
  run_stack(SF, 0, 24, Bn, tok_f, resid, Bf, stream);
  k_rmsnorm<<<Bn * SF.L, 32, 0, stream>>>(tok_f, resid, norm_f_fast, fastfeat,
                                          Bn * SF.L, SF.d);

  // --- slow stack: 12 layers, lateral, 12 layers ---
  hipMemsetAsync(resid, 0, TOK * sizeof(float), stream);
  run_stack(SS, 0, 12, Bn, tok_s, resid, Bf, stream);
  k_lateral<<<gsz((long long)TOK, EB), EB, 0, stream>>>(
      tok_s, fastfeat, lat_w, lat_b, lat_bng, lat_bnb, lat_bnm, lat_bnv, lat_out, Bn);
  run_stack(SS, 12, 24, Bn, lat_out, resid, Bf, stream);
  k_rmsnorm<<<Bn * SS.L, 32, 0, stream>>>(lat_out, resid, norm_f_slow, slowfeat,
                                          Bn * SS.L, SS.d);

  // --- upsample conv -> mean pool -> head GEMM (WMMA) ---
  k_upconv<<<gsz((long long)Bn * 1568 * 192, EB), EB, 0, stream>>>(
      fastfeat, slowfeat, ups_w, ups_b, up_tok, Bn);
  k_meanpool<<<gsz(Bn * 192, EB), EB, 0, stream>>>(up_tok, pooled, Bn, 1568, 192);
  k_gemm_wmma<<<gtiles(Bn, 1000), 32, 0, stream>>>(
      pooled, 192, head_w, 192, head_b, (float*)d_out, 1000, Bn, 1000, 192);
}
